// dfs_20109036880320
// MI455X (gfx1250) — matrix operand order repaired and checked by reading, not compiled, not run on hardware
//
#include <hip/hip_runtime.h>
#include <hip/hip_bf16.h>
#include <math.h>

typedef __attribute__((ext_vector_type(16))) _Float16 v16h;
typedef __attribute__((ext_vector_type(8)))  _Float16 v8h;
typedef __attribute__((ext_vector_type(8)))  float    v8f;

#define NB   8
#define CC   64
#define HH   128
#define WW   128
#define GK2  72
#define K2   9
#define EPSF 1e-6f
#define LSTR 136   // halves; [64][LSTR] tiles, 272B rows (16B aligned)
#define CSTR 72    // halves; [128][CSTR] transposed q tiles, 144B rows
#define ASTR 136   // halves; [128][ASTR] att tile

// ---------------------------------------------------------------------------
// WMMA 16x16x32 f16 fragment index maps (CDNA5 ISA 7.12.2, wave32)
// A (16x32): lane -> m (lane%16); halves 0-7 -> K (lhi*8 + e),
//            halves 8-15 -> K (16 + lhi*8 + e-8)   [+32 per k-step]
// B (32x16): lane -> n (lane%16); halves 0-15 -> K same as A (kmapA)
// C/D (16x16 f32): VGPR r -> m = r + 8*lhi ; lane%16 -> n
// ---------------------------------------------------------------------------
__device__ __forceinline__ int kmapA(int e, int lhi) {
    return ((e & 8) << 1) + (lhi ? 8 : 0) + (e & 7);
}
__device__ __forceinline__ int kmapB(int e, int lhi) {
    return e + (lhi ? 16 : 0);
}
__device__ __forceinline__ v16h cat8(v8h lo, v8h hi) {
    return __builtin_shufflevector(lo, hi, 0,1,2,3,4,5,6,7,8,9,10,11,12,13,14,15);
}

// ---------------------------------------------------------------------------
// Kernel 1: global average pool  x[n,c,:,:] -> pooled[n*64+c]
// ---------------------------------------------------------------------------
__global__ __launch_bounds__(256) void pool_kernel(const float* __restrict__ x,
                                                   float* __restrict__ pooled) {
    __shared__ float red[256];
    const int nc = blockIdx.x;
    const float* b = x + (size_t)nc * (HH * WW);
    float s = 0.f;
    for (int i = threadIdx.x; i < HH * WW; i += 256) s += b[i];
    red[threadIdx.x] = s;
    __syncthreads();
    for (int o = 128; o > 0; o >>= 1) {
        if (threadIdx.x < o) red[threadIdx.x] += red[threadIdx.x + o];
        __syncthreads();
    }
    if (threadIdx.x == 0) pooled[nc] = red[0] * (1.f / (HH * WW));
}

// ---------------------------------------------------------------------------
// Kernel 2: lf = pooled @ conv_w^T ; LayerNorm(72) ; softmax per group of 9
// ---------------------------------------------------------------------------
__global__ __launch_bounds__(128) void filt_kernel(const float* __restrict__ pooled,
                                                   const float* __restrict__ conv_w,
                                                   const float* __restrict__ ln_w,
                                                   const float* __restrict__ ln_b,
                                                   float* __restrict__ filt) {
    __shared__ float pool[CC];
    __shared__ float lf[GK2];
    __shared__ float lfn[GK2];
    const int n = blockIdx.x;
    const int t = threadIdx.x;
    if (t < CC) pool[t] = pooled[n * CC + t];
    __syncthreads();
    if (t < GK2) {
        float a = 0.f;
        const float* wr = conv_w + t * CC;
        for (int c = 0; c < CC; ++c) a += pool[c] * wr[c];
        lf[t] = a;
    }
    __syncthreads();
    if (t < GK2) {
        float m = 0.f;
        for (int j = 0; j < GK2; ++j) m += lf[j];
        m *= (1.f / GK2);
        float v = 0.f;
        for (int j = 0; j < GK2; ++j) { float d = lf[j] - m; v += d * d; }
        v *= (1.f / GK2);
        lfn[t] = (lf[t] - m) * rsqrtf(v + EPSF) * ln_w[t] + ln_b[t];
    }
    __syncthreads();
    if (t < GK2) {
        const int g = t / K2;
        const float* gb = lfn + g * K2;
        float mx = gb[0];
        for (int j = 1; j < K2; ++j) mx = fmaxf(mx, gb[j]);
        float s = 0.f;
        for (int j = 0; j < K2; ++j) s += __expf(gb[j] - mx);
        filt[n * GK2 + t] = __expf(lfn[t] - mx) / s;
    }
}

// ---------------------------------------------------------------------------
// Fused 1x1-conv GEMM: dst = W @ (optional LN(src)) + bias
// src: [64][LSTR] f16 LDS; dst: TR ? [128][CSTR] (transposed) : [64][LSTR]
// Template params keep LDS address space inferable and kill runtime branches.
// ---------------------------------------------------------------------------
template <bool DO_LN, bool TR>
__device__ __forceinline__ void conv_gemm(const float* __restrict__ Wt,
                                          const float* __restrict__ Bb,
                                          const _Float16* src, _Float16* dst,
                                          const float* mean, const float* rstd,
                                          const float* __restrict__ gw,
                                          const float* __restrict__ gb,
                                          int wave, int llo, int lhi) {
    for (int tile = wave; tile < 32; tile += 8) {
        const int mt = tile >> 3, nt = tile & 7;
        const int w = nt * 16 + llo;
        float mw = 0.f, rw = 0.f;
        if (DO_LN) { mw = mean[w]; rw = rstd[w]; }
        v8f acc = {};
        #pragma unroll
        for (int ks = 0; ks < 2; ++ks) {
            v16h A, B;
            #pragma unroll
            for (int e = 0; e < 16; ++e) {
                const int kA = ks * 32 + kmapA(e, lhi);
                A[e] = (_Float16)Wt[(mt * 16 + llo) * CC + kA];
                const int c = ks * 32 + kmapA(e, lhi);
                float v = (float)src[c * LSTR + w];
                if (DO_LN) v = (v - mw) * rw * gw[c] + gb[c];
                B[e] = (_Float16)v;
            }
            acc = __builtin_amdgcn_wmma_f32_16x16x32_f16(
                      false, A, false, B, (short)0, acc, false, false);
        }
        if (TR) {
            v8h p;
            #pragma unroll
            for (int r = 0; r < 8; ++r)
                p[r] = (_Float16)(acc[r] + Bb[mt * 16 + r + lhi * 8]);
            *(v8h*)(dst + w * CSTR + mt * 16 + lhi * 8) = p;   // 16B aligned
        } else {
            #pragma unroll
            for (int r = 0; r < 8; ++r) {
                const int o = mt * 16 + r + lhi * 8;
                dst[o * LSTR + w] = (_Float16)(acc[r] + Bb[o]);
            }
        }
    }
}

// ---------------------------------------------------------------------------
// Kernel 3 (fused, one block = one (n,h) row)
// ---------------------------------------------------------------------------
__global__ __launch_bounds__(256) void fcam_kernel(
        const float* __restrict__ x,
        const float* __restrict__ filt,
        const float* __restrict__ nl_w, const float* __restrict__ nl_b,
        const float* __restrict__ nr_w, const float* __restrict__ nr_b,
        const float* __restrict__ lp1_w, const float* __restrict__ lp1_b,
        const float* __restrict__ rp1_w, const float* __restrict__ rp1_b,
        const float* __restrict__ lp2_w, const float* __restrict__ lp2_b,
        const float* __restrict__ rp2_w, const float* __restrict__ rp2_b,
        const float* __restrict__ beta, const float* __restrict__ gamma,
        float* __restrict__ out) {
    extern __shared__ char smem[];
    _Float16* lowS  = (_Float16*)smem;            // [64][LSTR]
    _Float16* highS = lowS  + CC * LSTR;
    _Float16* vlS   = highS + CC * LSTR;
    _Float16* vhS   = vlS   + CC * LSTR;
    _Float16* qlS   = vhS   + CC * LSTR;          // [128][CSTR] (w-major)
    _Float16* qhS   = qlS   + HH * CSTR;
    _Float16* attS  = qhS   + HH * CSTR;          // [128][ASTR]
    float* stats    = (float*)(attS + HH * ASTR); // 8 * 128 floats
    float* meanL  = stats;        float* rstdL  = stats + 128;
    float* meanH  = stats + 256;  float* rstdH  = stats + 384;
    float* rowMax = stats + 512;  float* rowInv = stats + 640;
    float* colMax = stats + 768;  float* colInv = stats + 896;

    const int h    = blockIdx.x;
    const int n0   = blockIdx.y;
    const int t    = threadIdx.x;
    const int lane = t & 31;
    const int wave = t >> 5;
    const int llo  = lane & 15;
    const int lhi  = (lane >> 4) & 1;

    // ------------- Phase A: dynamic low-pass, high-pass, LN stats ----------
    {
        float* pscr = (float*)attS;               // att region as scratch
        const int w    = t & 127;
        const int half = t >> 7;
        const int wm = (w == 0)      ? 1      : w - 1;   // reflect pad
        const int wp = (w == WW - 1) ? WW - 2 : w + 1;
        const int h0 = (h == 0)      ? 1      : h - 1;
        const int h2 = (h == HH - 1) ? HH - 2 : h + 1;
        {   // CDNA5 global_prefetch_b8 on the first stencil rows
            const float* xb0 = x + ((size_t)(n0 * CC + half * 32) * HH) * WW;
            __builtin_prefetch(xb0 + h0 * WW + w, 0, 3);
            __builtin_prefetch(xb0 + h  * WW + w, 0, 3);
            __builtin_prefetch(xb0 + h2 * WW + w, 0, 3);
        }
        float sL = 0.f, sL2 = 0.f, sH = 0.f, sH2 = 0.f;
        float fw[9];
        for (int ci = 0; ci < 32; ++ci) {
            const int c = half * 32 + ci;
            if ((c & 7) == 0) {
                const float* fr = filt + n0 * GK2 + (c >> 3) * K2;
                #pragma unroll
                for (int j = 0; j < 9; ++j) fw[j] = fr[j];
            }
            const float* xb = x + ((size_t)(n0 * CC + c) * HH) * WW;
            const float* r0 = xb + h0 * WW;
            const float* r1 = xb + h  * WW;
            const float* r2 = xb + h2 * WW;
            float lo = fw[0]*r0[wm] + fw[1]*r0[w] + fw[2]*r0[wp]
                     + fw[3]*r1[wm] + fw[4]*r1[w] + fw[5]*r1[wp]
                     + fw[6]*r2[wm] + fw[7]*r2[w] + fw[8]*r2[wp];
            float hi = r1[w] - lo;
            lowS [c * LSTR + w] = (_Float16)lo;
            highS[c * LSTR + w] = (_Float16)hi;
            sL += lo; sL2 += lo * lo; sH += hi; sH2 += hi * hi;
        }
        pscr[half * 512 +   0 + w] = sL;
        pscr[half * 512 + 128 + w] = sL2;
        pscr[half * 512 + 256 + w] = sH;
        pscr[half * 512 + 384 + w] = sH2;
    }
    __syncthreads();
    if (t < 128) {
        float* pscr = (float*)attS;
        const int w = t;
        float sL  = pscr[w]       + pscr[512 + w];
        float sL2 = pscr[128 + w] + pscr[640 + w];
        float sH  = pscr[256 + w] + pscr[768 + w];
        float sH2 = pscr[384 + w] + pscr[896 + w];
        float mL = sL * (1.f / CC), mH = sH * (1.f / CC);
        meanL[w] = mL; rstdL[w] = rsqrtf(sL2 * (1.f / CC) - mL * mL + EPSF);
        meanH[w] = mH; rstdH[w] = rsqrtf(sH2 * (1.f / CC) - mH * mH + EPSF);
    }
    __syncthreads();

    // ------------- Phase B: four 64x128x64 GEMMs (ql,qh,vl,vh) -------------
    conv_gemm<true,  true >(lp1_w, lp1_b, lowS,  qlS, meanL, rstdL, nl_w, nl_b, wave, llo, lhi);
    conv_gemm<true,  true >(rp1_w, rp1_b, highS, qhS, meanH, rstdH, nr_w, nr_b, wave, llo, lhi);
    conv_gemm<false, false>(lp2_w, lp2_b, lowS,  vlS, meanL, rstdL, nl_w, nl_b, wave, llo, lhi);
    conv_gemm<false, false>(rp2_w, rp2_b, highS, vhS, meanH, rstdH, nr_w, nr_b, wave, llo, lhi);
    __syncthreads();

    // ------------- Phase C: att[i][j] = scale * sum_c ql[c,i] qh[c,j] ------
    const float scale = 0.125f;   // 64^-0.5
    for (int tile = wave; tile < 64; tile += 8) {
        const int it = tile >> 3, jt = tile & 7;
        const int i = it * 16 + llo;
        const int j = jt * 16 + llo;
        v8f acc = {};
        #pragma unroll
        for (int ks = 0; ks < 2; ++ks) {
            const _Float16* ap = qlS + i * CSTR + ks * 32 + lhi * 8;
            v16h A = cat8(*(const v8h*)ap, *(const v8h*)(ap + 16));
            const _Float16* bp = qhS + j * CSTR + ks * 32 + lhi * 8;
            v16h B = cat8(*(const v8h*)bp, *(const v8h*)(bp + 16));
            acc = __builtin_amdgcn_wmma_f32_16x16x32_f16(
                      false, A, false, B, (short)0, acc, false, false);
        }
        #pragma unroll
        for (int r = 0; r < 8; ++r)
            attS[(it * 16 + r + lhi * 8) * ASTR + j] = (_Float16)(acc[r] * scale);
    }
    __syncthreads();

    // ------------- Phase D: softmax stats (rows for P1, cols for P2) -------
    if (t < 128) {
        const int i = t;
        float m = -1e30f;
        for (int j = 0; j < WW; ++j) m = fmaxf(m, (float)attS[i * ASTR + j]);
        float s = 0.f;
        for (int j = 0; j < WW; ++j) s += __expf((float)attS[i * ASTR + j] - m);
        rowMax[i] = m; rowInv[i] = 1.f / s;
    } else {
        const int j = t - 128;
        float m = -1e30f;
        for (int i = 0; i < WW; ++i) m = fmaxf(m, (float)attS[i * ASTR + j]);
        float s = 0.f;
        for (int i = 0; i < WW; ++i) s += __expf((float)attS[i * ASTR + j] - m);
        colMax[j] = m; colInv[j] = 1.f / s;
    }
    __syncthreads();

    // ------------- Phase E: f_h2l = P1 @ vh ; f_l2h = P2-apply vl ----------
    for (int tile = wave; tile < 32; tile += 8) {
        const int ct = tile >> 3, wt = tile & 7;
        const int cA = ct * 16 + llo;
        const int wn = wt * 16 + llo;
        const float rm = rowMax[wn], ri = rowInv[wn];
        const float cm = colMax[wn], cv = colInv[wn];
        v8f acc1 = {}, acc2 = {};
        #pragma unroll
        for (int ks = 0; ks < 4; ++ks) {
            const _Float16* a1p = vhS + cA * LSTR + ks * 32 + lhi * 8;
            v16h A1 = cat8(*(const v8h*)a1p, *(const v8h*)(a1p + 16));
            const _Float16* a2p = vlS + cA * LSTR + ks * 32 + lhi * 8;
            v16h A2 = cat8(*(const v8h*)a2p, *(const v8h*)(a2p + 16));
            const _Float16* b1p = attS + wn * ASTR + ks * 32 + lhi * 8;
            v16h Braw = cat8(*(const v8h*)b1p, *(const v8h*)(b1p + 16));
            v16h B1, B2;
            #pragma unroll
            for (int e = 0; e < 16; ++e) {
                // P1[i=wn, j] row-softmax, contiguous att row
                B1[e] = (_Float16)(__expf((float)Braw[e] - rm) * ri);
                // P2[i, j=wn] col-softmax, strided att column
                const int kB = ks * 32 + kmapA(e, lhi);
                B2[e] = (_Float16)(__expf((float)attS[kB * ASTR + wn] - cm) * cv);
            }
            acc1 = __builtin_amdgcn_wmma_f32_16x16x32_f16(
                       false, A1, false, B1, (short)0, acc1, false, false);
            acc2 = __builtin_amdgcn_wmma_f32_16x16x32_f16(
                       false, A2, false, B2, (short)0, acc2, false, false);
        }
        #pragma unroll
        for (int r = 0; r < 8; ++r) {
            const int c = ct * 16 + r + lhi * 8;
            out[(((size_t)n0 * CC + c) * HH + h) * WW + wn] =
                beta[c] * acc1[r] + gamma[c] * acc2[r];
        }
    }
}

// ---------------------------------------------------------------------------
extern "C" void kernel_launch(void* const* d_in, const int* in_sizes, int n_in,
                              void* d_out, int out_size, void* d_ws, size_t ws_size,
                              hipStream_t stream) {
    (void)in_sizes; (void)n_in; (void)out_size; (void)ws_size;
    const float* x      = (const float*)d_in[0];
    const float* conv_w = (const float*)d_in[1];
    const float* ln_w   = (const float*)d_in[2];
    const float* ln_b   = (const float*)d_in[3];
    const float* nl_w   = (const float*)d_in[4];
    const float* nl_b   = (const float*)d_in[5];
    const float* nr_w   = (const float*)d_in[6];
    const float* nr_b   = (const float*)d_in[7];
    const float* lp1_w  = (const float*)d_in[8];
    const float* lp1_b  = (const float*)d_in[9];
    const float* rp1_w  = (const float*)d_in[10];
    const float* rp1_b  = (const float*)d_in[11];
    const float* lp2_w  = (const float*)d_in[12];
    const float* lp2_b  = (const float*)d_in[13];
    const float* rp2_w  = (const float*)d_in[14];
    const float* rp2_b  = (const float*)d_in[15];
    const float* beta   = (const float*)d_in[16];
    const float* gamma  = (const float*)d_in[17];
    float* out = (float*)d_out;

    float* pooled = (float*)d_ws;      // 512 f32
    float* filt   = pooled + 512;      // 576 f32

    pool_kernel<<<NB * CC, 256, 0, stream>>>(x, pooled);
    filt_kernel<<<NB, 128, 0, stream>>>(pooled, conv_w, ln_w, ln_b, filt);

    const size_t smem = (size_t)4 * CC * LSTR * sizeof(_Float16)    // low/high/vl/vh
                      + (size_t)2 * HH * CSTR * sizeof(_Float16)    // ql/qh (transposed)
                      + (size_t)HH * ASTR * sizeof(_Float16)        // att
                      + (size_t)8 * 128 * sizeof(float);            // stats = 145408 B
    hipFuncSetAttribute((const void*)fcam_kernel,
                        hipFuncAttributeMaxDynamicSharedMemorySize, (int)smem);
    dim3 grid(HH, NB);
    fcam_kernel<<<grid, 256, smem, stream>>>(x, filt, nl_w, nl_b, nr_w, nr_b,
        lp1_w, lp1_b, rp1_w, rp1_b, lp2_w, lp2_b, rp2_w, rp2_b,
        beta, gamma, out);
}
